// rbig_block_69140383531343
// MI455X (gfx1250) — compile-verified, matrix operand order checked by reading, not run on hardware
//
#include <hip/hip_runtime.h>

typedef __attribute__((ext_vector_type(16))) _Float16 v16h;
typedef __attribute__((ext_vector_type(8)))  float    v8f;

#define D_DIM 784
#define D_PAD 800   // K-dim padding for Y / Vn columns (multiple of 32)
#define R_PAD 832   // reflection-dim padding for Vn rows / VnT columns (multiple of 64)
#define N_PTS 256
#define B_ROWS 128
#define HS_MIN_F 1e-7f
#define LOG_HS_MIN_F -16.11809565095832f
#define MASK_BOUND_F 5e-8f
#define HALF_LOG_2PI 0.9189385332046727f

// ---------------------------------------------------------------- erfinv/ndtri
__device__ __forceinline__ float erfinv_f(float x) {
  float w = -logf((1.0f - x) * (1.0f + x));
  float p;
  if (w < 5.0f) {
    w -= 2.5f;
    p = 2.81022636e-08f;
    p = fmaf(p, w, 3.43273939e-07f);
    p = fmaf(p, w, -3.5233877e-06f);
    p = fmaf(p, w, -4.39150654e-06f);
    p = fmaf(p, w, 0.00021858087f);
    p = fmaf(p, w, -0.00125372503f);
    p = fmaf(p, w, -0.00417768164f);
    p = fmaf(p, w, 0.246640727f);
    p = fmaf(p, w, 1.50140941f);
  } else {
    w = sqrtf(w) - 3.0f;
    p = -0.000200214257f;
    p = fmaf(p, w, 0.000100950558f);
    p = fmaf(p, w, 0.00134934322f);
    p = fmaf(p, w, -0.00367342844f);
    p = fmaf(p, w, 0.00573950773f);
    p = fmaf(p, w, -0.0076224613f);
    p = fmaf(p, w, 0.00943887047f);
    p = fmaf(p, w, 1.00167406f);
    p = fmaf(p, w, 2.83297682f);
  }
  return p * x;
}
__device__ __forceinline__ float ndtri_f(float p) {
  return 1.41421356237309505f * erfinv_f(2.0f * p - 1.0f);
}

// ---------------------------------------------------------------- zero scratch
__global__ void zero_kernel(float* __restrict__ p, int n) {
  const int i = blockIdx.x * blockDim.x + threadIdx.x;
  if (i < n) p[i] = 0.0f;
}

// ---------------------------------------------------------------- KDE kernel
// grid = 784 (one block per dimension d), block = 128 (one thread per batch b)
__global__ void kde_kernel(const float* __restrict__ x,
                           const float* __restrict__ dp,
                           const float* __restrict__ lhs,
                           const float* __restrict__ kw,
                           float* __restrict__ Y,        // [128][D_PAD]
                           float* __restrict__ contrib)  // [128][784]
{
  const int d = blockIdx.x;
  const int b = threadIdx.x;
  __shared__ float s_dp[N_PTS], s_lhs[N_PTS], s_kw[N_PTS];
  __shared__ float red[B_ROWS];

  for (int n = b; n < N_PTS; n += B_ROWS) {
    s_dp[n]  = dp[n * D_DIM + d];
    s_lhs[n] = lhs[n * D_DIM + d];
    s_kw[n]  = kw[n * D_DIM + d];
  }
  __syncthreads();

  // logsumexp of kde_weights over n (column d), fixed-order tree reductions
  float lm = -__builtin_inff();
  for (int n = b; n < N_PTS; n += B_ROWS) lm = fmaxf(lm, s_kw[n]);
  red[b] = lm; __syncthreads();
  for (int s = 64; s > 0; s >>= 1) {
    if (b < s) red[b] = fmaxf(red[b], red[b + s]);
    __syncthreads();
  }
  const float kmax = red[0];
  __syncthreads();
  float ssum = 0.0f;
  for (int n = b; n < N_PTS; n += B_ROWS) ssum += expf(s_kw[n] - kmax);
  red[b] = ssum; __syncthreads();
  for (int s = 64; s > 0; s >>= 1) {
    if (b < s) red[b] += red[b + s];
    __syncthreads();
  }
  const float lse_kw = kmax + logf(red[0]);

  const float xv = x[b * D_DIM + d];

  float m1 = -__builtin_inff(), s1 = 0.0f;
  float m2 = -__builtin_inff(), s2 = 0.0f;
  float m3 = -__builtin_inff(), s3 = 0.0f;
  auto upd = [](float& m, float& s, float t) {
    if (t > m) { s = s * expf(m - t) + 1.0f; m = t; }
    else       { s += expf(t - m); }
  };

  for (int n = 0; n < N_PTS; ++n) {
    const float lh  = s_lhs[n];
    const float h   = fmaxf(expf(lh), HS_MIN_F);
    const float lhc = fmaxf(lh, LOG_HS_MIN_F);
    const float u   = -(xv - s_dp[n]) / h;
    const float sp  = (u > 0.0f) ? (u + log1pf(expf(-u))) : log1pf(expf(u));
    const float w   = s_kw[n] - lse_kw;
    upd(m1, s1, -sp + w);                 // log_cdf stream
    upd(m2, s2, u - sp + w);              // log_sf stream
    upd(m3, s3, u - lhc - 2.0f * sp + w); // log_pdf stream
  }
  const float log_cdf = m1 + logf(s1);
  const float log_sf  = m2 + logf(s2);
  const float log_pdf = m3 + logf(s3);

  const float cdf = expf(log_cdf);
  const bool good  = (cdf > MASK_BOUND_F) && (cdf < 1.0f - MASK_BOUND_F);
  const bool right = (cdf >= 1.0f - MASK_BOUND_F);
  const bool left  = (cdf <= MASK_BOUND_F);

  float inv = ndtri_f(good ? cdf : 0.5f);
  const float badR = right ? log_sf  : -1.0f;
  const float badL = left  ? log_cdf : -1.0f;
  if (right) inv += sqrtf(-2.0f * badR);
  if (left)  inv -= sqrtf(-2.0f * badL);

  const float log_norm = -0.5f * inv * inv - HALF_LOG_2PI;
  float lgd = 0.0f;
  if (good)  lgd += log_norm;
  if (left)  lgd += 0.5f * logf(-2.0f * badL) - log_cdf;
  if (right) lgd += 0.5f * logf(-2.0f * badR) - log_sf;

  Y[b * D_PAD + d]       = inv;
  contrib[b * D_DIM + d] = log_pdf - lgd;
}

// ---------------------------------------------------------------- log_det sum
__global__ void logdet_kernel(const float* __restrict__ ld_in,
                              const float* __restrict__ contrib,
                              float* __restrict__ out_ld) {
  const int b = blockIdx.x, t = threadIdx.x;
  __shared__ float red[256];
  float s = 0.0f;
  for (int d = t; d < D_DIM; d += 256) s += contrib[b * D_DIM + d];
  red[t] = s; __syncthreads();
  for (int k = 128; k > 0; k >>= 1) {
    if (t < k) red[t] += red[t + k];
    __syncthreads();
  }
  if (t == 0) out_ld[b] = ld_in[b] + red[0];
}

// ---------------------------------------------------------------- normalize vs
// Vn[i][d] = vs[i][d] / ||vs[i]||, stored with row stride D_PAD (pads stay 0)
__global__ void normalize_kernel(const float* __restrict__ vs,
                                 float* __restrict__ Vn) {
  const int i = blockIdx.x, t = threadIdx.x;
  __shared__ float red[256];
  float s = 0.0f;
  for (int d = t; d < D_DIM; d += 256) { float v = vs[i * D_DIM + d]; s = fmaf(v, v, s); }
  red[t] = s; __syncthreads();
  for (int k = 128; k > 0; k >>= 1) {
    if (t < k) red[t] += red[t + k];
    __syncthreads();
  }
  const float inv = 1.0f / sqrtf(red[0]);
  for (int d = t; d < D_DIM; d += 256) Vn[i * D_PAD + d] = vs[i * D_DIM + d] * inv;
}

// ---------------------------------------------------------------- Vn transpose
// VnT[d][i] = Vn[i][d]; VnT row stride R_PAD (pad columns stay 0)
__global__ void transpose_kernel(const float* __restrict__ Vn,
                                 float* __restrict__ VnT) {
  const int idx = blockIdx.x * blockDim.x + threadIdx.x;
  if (idx < D_DIM * D_DIM) {
    const int d = idx / D_DIM, i = idx % D_DIM;
    VnT[d * R_PAD + i] = Vn[i * D_PAD + d];
  }
}

// ---------------------------------------------------------------- compact-WY T
// One workgroup. S = V^T V (kb x kb), T recurrence:
//   T[r][c] = -2 * sum_{q=r..c-1} T[r][q] * S[q][c],  T[c][c] = 2
// Emits T transposed (Tt[n][k] = T[k][n]) so the GEMM reads it N-x-K row-major.
__global__ void build_T_kernel(const float* __restrict__ Vn,
                               float* __restrict__ Tt,
                               int i0, int kb) {
  __shared__ float S[64 * 64];
  __shared__ float T[64 * 64];
  const int t = threadIdx.x; // 256 threads
  for (int p = t; p < 64 * 64; p += 256) T[p] = 0.0f;
  for (int p = t; p < kb * kb; p += 256) {
    const int i = p / kb, j = p % kb;
    const float* vi = Vn + (i0 + i) * D_PAD;
    const float* vj = Vn + (i0 + j) * D_PAD;
    float s = 0.0f;
    for (int d = 0; d < D_DIM; ++d) s = fmaf(vi[d], vj[d], s);
    S[i * 64 + j] = s;
  }
  __syncthreads();
  for (int c = 0; c < kb; ++c) {
    if (t < c) {
      float s = 0.0f;
      for (int q = t; q < c; ++q) s = fmaf(T[t * 64 + q], S[q * 64 + c], s);
      T[t * 64 + c] = -2.0f * s;
    }
    if (t == c) T[c * 64 + c] = 2.0f;
    __syncthreads();
  }
  for (int p = t; p < 64 * 64; p += 256) {
    const int r = p / 64, c = p % 64;
    Tt[p] = T[c * 64 + r]; // transpose on the way out
  }
}

// ---------------------------------------------------------------- WMMA GEMM
// ACCUM=false: C[M,N] =  A[M,K] * B^T          (B stored N x K row-major)
// ACCUM=true : C[M,N] -= A[M,K] * B^T
// One 16x16 C tile per wave32; grid supplies exactly tilesTotal waves, so EXEC
// is all-ones everywhere (WMMA requirement). M,N mult of 16; K mult of 32.
// ISA fragment layout: lane L holds A/B row (L&15); its 16 halves are the two
// contiguous 8-float runs at k = kk + 8*(L>=16) and k+16.
__device__ __forceinline__ v16h load_frag(const float* __restrict__ row, int kbase) {
  const float4* p = reinterpret_cast<const float4*>(row + kbase);
  const float4 c0 = p[0], c1 = p[1];   // k = kbase .. kbase+7
  const float4 c2 = p[4], c3 = p[5];   // k = kbase+16 .. kbase+23
  v16h f;
  f[0]  = (_Float16)c0.x; f[1]  = (_Float16)c0.y; f[2]  = (_Float16)c0.z; f[3]  = (_Float16)c0.w;
  f[4]  = (_Float16)c1.x; f[5]  = (_Float16)c1.y; f[6]  = (_Float16)c1.z; f[7]  = (_Float16)c1.w;
  f[8]  = (_Float16)c2.x; f[9]  = (_Float16)c2.y; f[10] = (_Float16)c2.z; f[11] = (_Float16)c2.w;
  f[12] = (_Float16)c3.x; f[13] = (_Float16)c3.y; f[14] = (_Float16)c3.z; f[15] = (_Float16)c3.w;
  return f;
}

template <bool ACCUM>
__global__ __launch_bounds__(128) void wmma_gemm_kernel(
    const float* __restrict__ A, int lda,
    const float* __restrict__ B, int ldb,
    float* __restrict__ C, int ldc,
    int tilesN, int K) {
  const int wave = threadIdx.x >> 5;
  const int lane = threadIdx.x & 31;
  const int tile = blockIdx.x * 4 + wave;   // grid sized exactly: no guard
  const int tm = (tile / tilesN) << 4;
  const int tn = (tile % tilesN) << 4;
  const int mn   = lane & 15;
  const int koff = (lane >> 4) << 3; // 0 or 8

  const float* arow = A + (tm + mn) * lda;
  const float* brow = B + (tn + mn) * ldb;

  v8f acc = {};
  for (int kk = 0; kk < K; kk += 32) {
    __builtin_prefetch(arow + kk + 32, 0, 3);
    __builtin_prefetch(brow + kk + 32, 0, 3);
    const v16h a = load_frag(arow, kk + koff);
    const v16h b = load_frag(brow, kk + koff);
    acc = __builtin_amdgcn_wmma_f32_16x16x32_f16(false, a, false, b,
                                                 (short)0, acc, false, false);
  }

  // Straight-line epilogue: per-lane column walk, 32-bit address math only.
  float* crow = C + (tm + ((lane >> 4) << 3)) * ldc + tn + (lane & 15);
#pragma unroll
  for (int r = 0; r < 8; ++r) {
    if (ACCUM) *crow -= acc[r];
    else       *crow  = acc[r];
    crow += ldc;
  }
}

// ---------------------------------------------------------------- copy out
__global__ void copy_kernel(const float* __restrict__ Y, float* __restrict__ out) {
  const int i = blockIdx.x * blockDim.x + threadIdx.x;
  if (i < B_ROWS * D_DIM) {
    const int b = i / D_DIM, d = i % D_DIM;
    out[i] = Y[b * D_PAD + d];
  }
}

// ---------------------------------------------------------------- launcher
extern "C" void kernel_launch(void* const* d_in, const int* in_sizes, int n_in,
                              void* d_out, int out_size, void* d_ws, size_t ws_size,
                              hipStream_t stream) {
  (void)in_sizes; (void)n_in; (void)out_size; (void)ws_size;
  const float* x       = (const float*)d_in[0];
  const float* log_det = (const float*)d_in[1];
  const float* dp      = (const float*)d_in[2];
  const float* lhs     = (const float*)d_in[3];
  const float* kw      = (const float*)d_in[4];
  const float* vs      = (const float*)d_in[5];
  float* out = (float*)d_out;

  // Workspace layout (bytes, 512-aligned):
  //   Y       : 128 x 800 f32            =  409600
  //   contrib : 128 x 784 f32            =  401408
  //   Vn      : 832 x 800 f32            = 2662400
  //   VnT     : 784 x 832 f32            = 2609152
  //   Pb, Zb  : 128 x 64 f32 each        =   32768 x2
  //   Tt      : 64 x 64 f32              =   16384
  char* ws = (char*)d_ws;
  float* Y       = (float*)(ws);
  float* contrib = (float*)(ws + 409600);
  float* Vn      = (float*)(ws + 811008);
  float* VnT     = (float*)(ws + 3473408);
  float* Pb      = (float*)(ws + 6082560);
  float* Zb      = (float*)(ws + 6115328);
  float* Tt      = (float*)(ws + 6148096);
  const int ws_floats = (6148096 + 16384) / 4;

  // 0) zero scratch (padding lanes must be exact zeros every call)
  zero_kernel<<<(ws_floats + 255) / 256, 256, 0, stream>>>((float*)ws, ws_floats);

  // 1) KDE gaussianization: Y = inverse_l (padded), contrib = log_pdf - lgd
  kde_kernel<<<D_DIM, B_ROWS, 0, stream>>>(x, dp, lhs, kw, Y, contrib);

  // 2) Normalized Householder vectors + transpose
  normalize_kernel<<<D_DIM, 256, 0, stream>>>(vs, Vn);
  transpose_kernel<<<(D_DIM * D_DIM + 255) / 256, 256, 0, stream>>>(Vn, VnT);

  // 3) log_det (independent of the rotation)
  logdet_kernel<<<B_ROWS, 256, 0, stream>>>(log_det, contrib, out + B_ROWS * D_DIM);

  // 4) Blocked compact-WY: Y <- Y (I - V T V^T), 13 blocks of <=64 reflections.
  //    All GEMMs run un-ragged (N=64 / K padded); zero pads make it exact.
  for (int i0 = 0; i0 < D_DIM; i0 += 64) {
    const int kb = (D_DIM - i0 < 64) ? (D_DIM - i0) : 64;
    build_T_kernel<<<1, 256, 0, stream>>>(Vn, Tt, i0, kb);

    // P[128,64] = Y[128,800] * Vn_block^T : 32 tiles = 8 blocks x 4 waves
    wmma_gemm_kernel<false><<<8, 128, 0, stream>>>(
        Y, D_PAD, Vn + i0 * D_PAD, D_PAD, Pb, 64, /*tilesN=*/4, /*K=*/D_PAD);

    // Z[128,64] = P * T : 32 tiles
    wmma_gemm_kernel<false><<<8, 128, 0, stream>>>(
        Pb, 64, Tt, 64, Zb, 64, /*tilesN=*/4, /*K=*/64);

    // Y[128,784] -= Z * V^T : 392 tiles = 98 blocks x 4 waves
    wmma_gemm_kernel<true><<<98, 128, 0, stream>>>(
        Zb, 64, VnT + i0, R_PAD, Y, D_PAD, /*tilesN=*/49, /*K=*/64);
  }

  // 5) x_out
  copy_kernel<<<(B_ROWS * D_DIM + 255) / 256, 256, 0, stream>>>(Y, out);
}
